// SimpleBlock_90718299226282
// MI455X (gfx1250) — matrix operand order repaired and checked by reading, not compiled, not run on hardware
//
#include <hip/hip_runtime.h>
#include <hip/hip_fp16.h>

typedef _Float16 v16h __attribute__((ext_vector_type(16)));
typedef _Float16 v8h  __attribute__((ext_vector_type(8)));
typedef float    v8f  __attribute__((ext_vector_type(8)));

#define KP_EXTENT   0.048f                 // 0.1*1.2/2.5
#define INV_EXTENT  (1.0f/0.048f)
#define BN_EPS      1e-5f
#define LEAKY       0.1f

#define NH    32      // neighbors per point == WMMA K-dim
#define NK    15      // kernel points (padded to 16 in GEMM1)
#define CIN   64
#define COUT  64
#define RTOT  960     // 15*64 flattened contraction dim of GEMM2
#define KT    30      // 960/32 K-tiles
#define NT    4       // 64/16 N-tiles

// ---- workspace byte offsets ----
#define WS_BFRAG  0         // 61440 halves = 122880 B (pre-swizzled B fragments)
#define WS_STATS  122880    // 128 floats: sum[64], sumsq[64]
#define WS_MINV   123392    // 128 floats: mean[64], inv_std[64]

// ---- LDS byte offsets (dynamic shared) ----
#define LDS_B      0        // 122880 B : B fragments (shared by all 4 waves)
#define LDS_CSUM   122880   // 64 f32
#define LDS_CSQ    123136   // 64 f32
#define LDS_WAVE   123392   // per-wave staging base
#define WAVE_BYTES 35840    // wlds 1024 + xgT 4096 + wtile 30720
#define SMEM_TOTAL (LDS_WAVE + 4*WAVE_BYTES)   // 266752 B < 320 KB

// Build a 16-bit A-fragment in contiguous K order (same order as B):
// lanes 0-15 hold K = 0..15 (a0=0), lanes 16-31 hold K = 16..31 (a0=16).
static __device__ __forceinline__ v16h load_fragA(const _Float16* base, int a0) {
  v8h lo = *(const v8h*)(base + a0);
  v8h hh = *(const v8h*)(base + a0 + 8);
  v16h r;
#pragma unroll
  for (int i = 0; i < 8; ++i) { r[i] = lo[i]; r[i + 8] = hh[i]; }
  return r;
}

// Pre-swizzle weights [15][64][64] f32 -> f16 B-fragments.
// B layout (16-bit, 32x16): lanes 0-15 hold K=0..15 of column n=lane,
// lanes 16-31 hold K=16..31 of column n=lane-16, halves in K order.
// Fragment e-th half for (kt,nt,lane): r = kt*32 + e + (lane>=16?16:0), n = nt*16 + (lane&15).
__global__ void prep_bfrag(const float* __restrict__ weights, _Float16* __restrict__ bfrag) {
  int tid = blockIdx.x * 256 + threadIdx.x;
  if (tid >= KT * NT * 32 * 16) return;
  int e  = tid & 15;
  int l  = (tid >> 4) & 31;
  int nt = (tid >> 9) & 3;
  int kt = tid >> 11;
  int r  = kt * 32 + e + ((l >= 16) ? 16 : 0);     // r = k*64 + c, r < 960
  int n  = nt * 16 + (l & 15);
  bfrag[tid] = (_Float16)weights[r * COUT + n];    // flat (k*64+c)*64 + d
}

__global__ void zero_stats(float* __restrict__ s) {
  if (threadIdx.x < 128) s[threadIdx.x] = 0.f;
}

extern "C" __global__ __launch_bounds__(128)
void kpconv_main(const float* __restrict__ q_pts, const float* __restrict__ s_pts,
                 const int* __restrict__ nbr, const float* __restrict__ x,
                 const float* __restrict__ kpts, const _Float16* __restrict__ bfrag,
                 float* __restrict__ out_raw, float* __restrict__ gstats, int N)
{
  extern __shared__ unsigned char smem[];
  _Float16* Blds = (_Float16*)(smem + LDS_B);
  float* csum = (float*)(smem + LDS_CSUM);
  float* csq  = (float*)(smem + LDS_CSQ);
  const int wave = threadIdx.x >> 5;
  const int lane = threadIdx.x & 31;
  unsigned char* wb = smem + LDS_WAVE + wave * WAVE_BYTES;
  _Float16* wlds  = (_Float16*)wb;             // [16][32]  Wt (kernel x neighbor)
  _Float16* xgT   = (_Float16*)(wb + 1024);    // [64][32]  X^T (channel x neighbor)
  _Float16* wtile = (_Float16*)(wb + 5120);    // [16][960] GEMM2 A staging

  { // cooperative copy of pre-swizzled B fragments into LDS (reused all tiles)
    const uint4* src = (const uint4*)bfrag;
    uint4* dst = (uint4*)Blds;
    for (int i = threadIdx.x; i < 122880 / 16; i += 128) dst[i] = src[i];
  }
  if (threadIdx.x < 64) { csum[threadIdx.x] = 0.f; csq[threadIdx.x] = 0.f; }

  float kpx[NK], kpy[NK], kpz[NK];    // uniform -> scalar loads
#pragma unroll
  for (int k = 0; k < NK; ++k) {
    kpx[k] = kpts[k * 3 + 0]; kpy[k] = kpts[k * 3 + 1]; kpz[k] = kpts[k * 3 + 2];
  }
  __syncthreads();

  const bool hiHalf = lane >= 16;
  const int lh = lane & 15;
  const int a0 = hiHalf ? 16 : 0;    // A-fragment K base
  const int b0 = hiHalf ? 16 : 0;    // B-fragment K base
  const int mBase = hiHalf ? 8 : 0;  // C/D row base

  const int n_tiles = (N + 15) >> 4;
  const int wgid  = blockIdx.x * 4 + wave;
  const int wstep = gridDim.x * 4;

  const v8f vzero = {0.f, 0.f, 0.f, 0.f, 0.f, 0.f, 0.f, 0.f};

  for (int t = wgid; t < n_tiles; t += wstep) {
    const int P = t << 4;
    __builtin_prefetch(nbr + (size_t)P * NH, 0, 1);   // global_prefetch_b8

    // ---- Phase 1: per point, stage influences/features then GEMM1 via WMMA ----
    for (int p = 0; p < 16; ++p) {
      const int n = P + p;
      int idx = -1;
      float rx = 1e6f, ry = 1e6f, rz = 1e6f;
      if (n < N) {
        idx = nbr[(size_t)n * NH + lane];             // lane == neighbor h
        if ((unsigned)idx < (unsigned)N) {
          float qx = q_pts[n * 3 + 0], qy = q_pts[n * 3 + 1], qz = q_pts[n * 3 + 2];
          rx = s_pts[idx * 3 + 0] - qx;
          ry = s_pts[idx * 3 + 1] - qy;
          rz = s_pts[idx * 3 + 2] - qz;
        } else idx = -1;
      }
#pragma unroll
      for (int k = 0; k < NK; ++k) {                  // linear influence weights
        float dx = rx - kpx[k], dy = ry - kpy[k], dz = rz - kpz[k];
        float d = sqrtf(dx * dx + dy * dy + dz * dz);
        float w = fmaxf(1.f - d * INV_EXTENT, 0.f);
        wlds[k * 32 + lane] = (_Float16)w;
      }
      wlds[NK * 32 + lane] = (_Float16)0.f;           // pad kernel row 15

      if (idx >= 0) {                                 // feature gather -> X^T staging
        const float4* xr = (const float4*)(x + (size_t)idx * CIN);
#pragma unroll
        for (int j = 0; j < 16; ++j) {
          float4 f = xr[j];
          xgT[(4 * j + 0) * 32 + lane] = (_Float16)f.x;
          xgT[(4 * j + 1) * 32 + lane] = (_Float16)f.y;
          xgT[(4 * j + 2) * 32 + lane] = (_Float16)f.z;
          xgT[(4 * j + 3) * 32 + lane] = (_Float16)f.w;
        }
      } else {
#pragma unroll
        for (int c = 0; c < CIN; ++c) xgT[c * 32 + lane] = (_Float16)0.f;
      }
      // wave-local LDS fence: staging writes (all lanes) -> fragment reads
      asm volatile("s_wait_dscnt 0" ::: "memory");

      // GEMM1: [16k x 32h] @ [32h x 64c], H==32 -> one WMMA per 16-channel tile.
      // Issue the 4 independent WMMAs back-to-back (shared A fragment) so the
      // WMMA->VALU D-read hazard slots are filled with useful work, then convert.
      v16h a = load_fragA(wlds + lh * 32, a0);
      v8f acc1[NT];
#pragma unroll
      for (int ct = 0; ct < NT; ++ct) {
        v16h b = *(const v16h*)(xgT + (ct * 16 + lh) * 32 + b0);
        acc1[ct] = __builtin_amdgcn_wmma_f32_16x16x32_f16(false, a, false, b,
                                                          (short)0, vzero, false, false);
      }
#pragma unroll
      for (int ct = 0; ct < NT; ++ct) {
#pragma unroll
        for (int r = 0; r < 8; ++r)
          wtile[p * RTOT + (r + mBase) * COUT + ct * 16 + lh] = (_Float16)acc1[ct][r];
      }
    }
    asm volatile("s_wait_dscnt 0" ::: "memory");

    // ---- Phase 2: batched GEMM2 [16 pts x 960] @ [960 x 64] ----
    // Per K-tile: one shared A fragment feeds 4 independent N-tile accumulators
    // (back-to-back independent WMMAs, A-fragment LDS traffic amortized 4x).
    v8f acc2[NT];
#pragma unroll
    for (int nt = 0; nt < NT; ++nt) acc2[nt] = vzero;
    for (int kt = 0; kt < KT; ++kt) {
      v16h a = load_fragA(wtile + lh * RTOT + kt * 32, a0);
#pragma unroll
      for (int nt = 0; nt < NT; ++nt) {
        v16h b = *(const v16h*)(Blds + ((kt * 4 + nt) * 32 + lane) * 16);
        acc2[nt] = __builtin_amdgcn_wmma_f32_16x16x32_f16(false, a, false, b,
                                                          (short)0, acc2[nt], false, false);
      }
    }
#pragma unroll
    for (int nt = 0; nt < NT; ++nt) {
      const int ch = nt * 16 + lh;
      float s1 = 0.f, s2 = 0.f;
#pragma unroll
      for (int r = 0; r < 8; ++r) {
        int pn = P + r + mBase;
        float v = acc2[nt][r];
        if (pn < N) { out_raw[(size_t)pn * COUT + ch] = v; s1 += v; s2 += v * v; }
      }
      atomicAdd(&csum[ch], s1);   // ds_add_f32
      atomicAdd(&csq[ch],  s2);
    }
  }
  __syncthreads();
  if (threadIdx.x < 64)       atomicAdd(&gstats[threadIdx.x], csum[threadIdx.x]);
  else if (threadIdx.x < 128) atomicAdd(&gstats[threadIdx.x], csq[threadIdx.x - 64]);
}

__global__ void finalize_stats(const float* __restrict__ gstats,
                               float* __restrict__ minv, int N) {
  int c = threadIdx.x;
  if (c < 64) {
    float invN = 1.f / (float)N;
    float mean = gstats[c] * invN;
    float var  = gstats[64 + c] * invN - mean * mean;
    minv[c]      = mean;
    minv[64 + c] = rsqrtf(fmaxf(var, 0.f) + BN_EPS);
  }
}

__global__ void norm_lrelu(float* __restrict__ out, const float* __restrict__ minv,
                           int total) {
  int i = (blockIdx.x * 256 + threadIdx.x) * 4;
  if (i >= total) return;
  float4 v = *(float4*)(out + i);
  int c = i & 63;
  float4 m  = *(const float4*)(minv + c);
  float4 iv = *(const float4*)(minv + 64 + c);
  v.x = (v.x - m.x) * iv.x; v.y = (v.y - m.y) * iv.y;
  v.z = (v.z - m.z) * iv.z; v.w = (v.w - m.w) * iv.w;
  v.x = v.x >= 0.f ? v.x : LEAKY * v.x;
  v.y = v.y >= 0.f ? v.y : LEAKY * v.y;
  v.z = v.z >= 0.f ? v.z : LEAKY * v.z;
  v.w = v.w >= 0.f ? v.w : LEAKY * v.w;
  *(float4*)(out + i) = v;
}

extern "C" void kernel_launch(void* const* d_in, const int* in_sizes, int n_in,
                              void* d_out, int out_size, void* d_ws, size_t ws_size,
                              hipStream_t stream) {
  const float* q_pts = (const float*)d_in[0];
  const float* s_pts = (const float*)d_in[1];
  const int*   nbr   = (const int*)d_in[2];
  const float* x     = (const float*)d_in[3];
  const float* kpts  = (const float*)d_in[4];
  const float* wgt   = (const float*)d_in[5];
  const int N = in_sizes[3] / CIN;

  float* out = (float*)d_out;
  unsigned char* ws = (unsigned char*)d_ws;
  _Float16* bfrag = (_Float16*)(ws + WS_BFRAG);
  float* gstats   = (float*)(ws + WS_STATS);
  float* minv     = (float*)(ws + WS_MINV);

  prep_bfrag<<<(KT * NT * 32 * 16 + 255) / 256, 256, 0, stream>>>(wgt, bfrag);
  zero_stats<<<1, 128, 0, stream>>>(gstats);

  hipFuncSetAttribute(reinterpret_cast<const void*>(kpconv_main),
                      hipFuncAttributeMaxDynamicSharedMemorySize, SMEM_TOTAL);
  kpconv_main<<<512, 128, SMEM_TOTAL, stream>>>(q_pts, s_pts, nbr, x, kpts,
                                                bfrag, out, gstats, N);

  finalize_stats<<<1, 64, 0, stream>>>(gstats, minv, N);

  const int total = N * COUT;
  norm_lrelu<<<(total / 4 + 255) / 256, 256, 0, stream>>>(out, minv, total);
}
